// NODE_pose_43628277792928
// MI455X (gfx1250) — compile-verified, matrix operand order checked by reading, not run on hardware
//
#include <hip/hip_runtime.h>
#include <math.h>

typedef __attribute__((ext_vector_type(16))) __bf16        v16bf;
typedef __attribute__((ext_vector_type(8)))  float         v8f;
typedef __attribute__((ext_vector_type(8)))  unsigned int  v8u;
typedef __attribute__((ext_vector_type(4)))  unsigned int  v4u;

static_assert(sizeof(v16bf) == 32, "v16bf must be 32 bytes");

#define NB 256
#define LL 100
#define CC 256
#define SZ (NB * CC * LL)   // 6,553,600 f32 elements

// ---------------- activations (compile-time selected) ----------------
template<int ACT>
__device__ __forceinline__ float act(float v) {
    if constexpr (ACT == 1) return v > 0.f ? v : 0.f;                  // relu
    else if constexpr (ACT == 2) return v > 0.f ? v : (expf(v) - 1.f); // elu
    else if constexpr (ACT == 3) return tanhf(v);                      // tanh
    else return v;
}

// ---------------- weight prep: ODE conv weights -> bf16 [3][256][256] ----------------
__global__ void prep_ode_w(const float* __restrict__ w, __bf16* __restrict__ Wk,
                           float* __restrict__ tvec) {
    int idx = blockIdx.x * 256 + threadIdx.x;
    if (idx >= 3 * 256 * 256) return;
    int kk = idx / (256 * 256);
    int rem = idx % (256 * 256);
    int co = rem / 256, ci = rem % 256;
    // w layout [co][257][3][3], take kh=1 row, input channel ci+1 (channel 0 is t)
    Wk[idx] = (__bf16)w[((co * 257 + (ci + 1)) * 3 + 1) * 3 + kk];
    if (ci == 0) tvec[kk * 256 + co] = w[((co * 257 + 0) * 3 + 1) * 3 + kk];
}

// wc1 weights [64][256][3][3] -> bf16 [3][64][256] (kh=1)
__global__ void prep_wc(const float* __restrict__ w, __bf16* __restrict__ Wk, int Cout) {
    int idx = blockIdx.x * 256 + threadIdx.x;
    if (idx >= 3 * Cout * 256) return;
    int kk = idx / (Cout * 256);
    int rem = idx % (Cout * 256);
    int co = rem / 256, ci = rem % 256;
    Wk[idx] = (__bf16)w[((co * 256 + ci) * 3 + 1) * 3 + kk];
}

// ---------------- plain small convolutions (frontend + tail) ----------------
// weight index = (co*Cin+ci)*kwt + koff + k ; 1d: kwt=3,koff=0 ; 2d mid-row: kwt=9,koff=3
template<int ACT>
__global__ void conv_plain(const float* __restrict__ in, const float* __restrict__ w,
                           const float* __restrict__ b, float* __restrict__ out,
                           int Cin, int Cout, int kwt, int koff) {
    int idx = blockIdx.x * 256 + threadIdx.x;
    int total = NB * Cout * LL;
    if (idx >= total) return;
    int l = idx % LL;
    int co = (idx / LL) % Cout;
    int n = idx / (LL * Cout);
    const float* ip = in + (size_t)n * Cin * LL;
    float s = b[co];
    for (int ci = 0; ci < Cin; ++ci) {
        const float* wr = w + ((size_t)co * Cin + ci) * kwt + koff;
        const float* xr = ip + ci * LL + l;
        if (l > 0) s += wr[0] * xr[-1];
        s += wr[1] * xr[0];
        if (l < LL - 1) s += wr[2] * xr[1];
    }
    out[idx] = act<ACT>(s);
}

// ---------------- batch norm over (N, L) per channel ----------------
__global__ void bn_stats(const float* __restrict__ in, const float* __restrict__ g,
                         const float* __restrict__ bb, float* __restrict__ scale,
                         float* __restrict__ shift, int C) {
    int c = blockIdx.x, tid = threadIdx.x;
    __shared__ float s1[256], s2[256];
    float a = 0.f, q = 0.f;
    for (int i = tid; i < NB * LL; i += 256) {
        int n = i / LL, l = i % LL;
        float v = in[((size_t)n * C + c) * LL + l];
        a += v; q += v * v;
    }
    s1[tid] = a; s2[tid] = q;
    __syncthreads();
    for (int st = 128; st; st >>= 1) {
        if (tid < st) { s1[tid] += s1[tid + st]; s2[tid] += s2[tid + st]; }
        __syncthreads();
    }
    if (tid == 0) {
        const float inv_n = 1.f / (float)(NB * LL);
        float m = s1[0] * inv_n;
        float v = s2[0] * inv_n - m * m;
        float sc = g[c] * rsqrtf(v + 1e-5f);
        scale[c] = sc;
        shift[c] = bb[c] - m * sc;
    }
}

template<int ACT>
__global__ void bn_apply(const float* __restrict__ in, const float* __restrict__ scale,
                         const float* __restrict__ shift, float* __restrict__ out, int C) {
    int idx = blockIdx.x * 256 + threadIdx.x;
    if (idx >= NB * C * LL) return;
    int c = (idx / LL) % C;
    out[idx] = act<ACT>(in[idx] * scale[c] + shift[c]);
}

// ---------------- group norm (32 groups of 8 ch x 100) ----------------
// TRANS=1: write bf16 transposed Xt[n][l][c]; TRANS=0: write f32 [n][c][l]
template<int ACT, int TRANS>
__global__ void group_norm(const float* __restrict__ in, const float* __restrict__ g,
                           const float* __restrict__ bb, float* __restrict__ of,
                           __bf16* __restrict__ ob) {
    int wv = threadIdx.x >> 5, lane = threadIdx.x & 31;
    int gid = blockIdx.x * 8 + wv;          // 0 .. NB*32-1
    int n = gid >> 5, grp = gid & 31;
    const float* base = in + ((size_t)n * CC + grp * 8) * LL;
    float s = 0.f, q = 0.f;
    for (int i = lane; i < 800; i += 32) { float v = base[i]; s += v; q += v * v; }
    for (int m = 16; m; m >>= 1) { s += __shfl_xor(s, m, 32); q += __shfl_xor(q, m, 32); }
    float mean = s * (1.f / 800.f);
    float inv = rsqrtf(q * (1.f / 800.f) - mean * mean + 1e-5f);
    for (int i = lane; i < 800; i += 32) {
        int c = grp * 8 + i / LL, l = i % LL;
        float v = (base[i] - mean) * inv * g[c] + bb[c];
        v = act<ACT>(v);
        if constexpr (TRANS) ob[((size_t)n * LL + l) * CC + c] = (__bf16)v;
        else                 of[((size_t)n * CC + c) * LL + l] = v;
    }
}

// ---------------- WMMA conv (block-cooperative, async-LDS-staged B) ----------------
// O[n][co][l] = sum_k W_k (Cout x CC) * X shifted by k-1.
// Xt: bf16 [n][100][256] ; Wk: bf16 [3][Cout][256] ; tvec (may be null): f32 [3][Cout]
// One block per (l-tile, sample): COT waves, wave ct owns co-tile ct.
// 18 activation rows (16 + halo) staged into LDS via GLOBAL_LOAD_ASYNC_TO_LDS_B128
// (ASYNCcnt-tracked, no VGPR round-trip); boundary rows zero-filled with ds_store.
// B fragments then come from ds_load_b128 — no divergent masking in the GEMM loop.
template<int ACT, int COT>   // COT: co-tiles per block (16 -> Cout=256, 4 -> Cout=64)
__global__ void __launch_bounds__(COT * 32)
wmma_conv(const __bf16* __restrict__ Xt, const __bf16* __restrict__ Wk,
          const float* __restrict__ bias, const float* __restrict__ tvec,
          float t, float* __restrict__ out) {
    constexpr int Cout = COT * 16;
    const int lt = blockIdx.x;             // l tile (0..6)
    const int n  = blockIdx.y;             // sample
    const int tid = threadIdx.x;
    const int ct = tid >> 5;               // wave id == co tile
    const int lane = tid & 31;
    const int half = lane >> 4;
    const int r16 = lane & 15;
    const int kb = half * 8;               // K-subset base per ISA A/B fragment layout

    // ---- cooperative stage: rows [lt*16-1, lt*16+16] of Xt -> LDS, zero-padded ----
    __shared__ __align__(16) __bf16 sB[18 * CC];
    {
        const __bf16* xs = Xt + (size_t)n * LL * CC;
        for (int i = tid; i < 18 * 32; i += COT * 32) {   // 18 rows * 32 x 16B chunks
            int r = i >> 5, c4 = i & 31;
            int lrow = lt * 16 + r - 1;
            __bf16* ldst = sB + (size_t)i * 8;            // 16 bytes per chunk
            if (lrow >= 0 && lrow < LL) {
                const __bf16* gsrc = xs + (size_t)lrow * CC + c4 * 8;
                unsigned loff = (unsigned)(size_t)ldst;   // LDS byte address (low 32 bits)
                asm volatile("global_load_async_to_lds_b128 %0, %1, off"
                             :: "v"(loff), "v"(gsrc) : "memory");
            } else {
                *(v4u*)ldst = (v4u)0u;                    // halo rows: zeros via ds_store
            }
        }
    }
    asm volatile("s_wait_asynccnt 0x0" ::: "memory");     // our async writes landed in LDS
    __syncthreads();                                      // all waves' stages visible

    v8f acc = {0.f, 0.f, 0.f, 0.f, 0.f, 0.f, 0.f, 0.f};
#pragma unroll
    for (int kk = 0; kk < 3; ++kk) {
        const __bf16* ar = Wk + ((size_t)kk * Cout + (ct * 16 + r16)) * CC;
        if (kk < 2)   // hint next tap's weight rows toward the caches (global_prefetch_b8)
            __builtin_prefetch(Wk + ((size_t)(kk + 1) * Cout + (ct * 16 + r16)) * CC, 0, 1);
        const __bf16* brow = sB + (r16 + kk) * CC;   // staged row for this lane's B column
#pragma unroll
        for (int q = 0; q < 8; ++q) {
            const int cb = q * 32 + kb;
            v8u araw, braw;
            araw.lo = *(const v4u*)(ar + cb);        // K = cb .. cb+7
            araw.hi = *(const v4u*)(ar + cb + 16);   // K = cb+16 .. cb+23
            braw.lo = *(const v4u*)(brow + cb);      // ds_load_b128
            braw.hi = *(const v4u*)(brow + cb + 16);
            acc = __builtin_amdgcn_wmma_f32_16x16x32_bf16(
                      false, __builtin_bit_cast(v16bf, araw),
                      false, __builtin_bit_cast(v16bf, braw),
                      (short)0, acc, false, false);
        }
    }

    const int l = lt * 16 + r16;                     // D: N = lane&15
    if (l < LL) {
#pragma unroll
        for (int r = 0; r < 8; ++r) {
            int co = ct * 16 + half * 8 + r;         // D: M = r + 8*half
            float v = acc[r] + bias[co];
            if (tvec) {                              // t-channel closed form
                float tv = tvec[Cout + co];          // center tap always valid
                if (l > 0)      tv += tvec[co];
                if (l < LL - 1) tv += tvec[2 * Cout + co];
                v += t * tv;
            }
            out[((size_t)n * Cout + co) * LL + l] = act<ACT>(v);
        }
    }
}

// ---------------- elementwise RK4 combine: out = a*x + b*y + c*z ----------------
__global__ void comb3(float* out, const float* x, float a, const float* y, float b,
                      const float* z, float c, int total) {
    int idx = blockIdx.x * 256 + threadIdx.x;
    if (idx >= total) return;
    out[idx] = a * x[idx] + b * y[idx] + c * z[idx];
}

// ---------------- final head: 100->64->3->1 per (n, j) ----------------
__global__ void head(const float* __restrict__ h, const float* __restrict__ wl1,
                     const float* __restrict__ bl1, const float* __restrict__ wl2,
                     const float* __restrict__ bl2, const float* __restrict__ wl3,
                     const float* __restrict__ bl3, float* __restrict__ out) {
    int idx = blockIdx.x * 256 + threadIdx.x;
    if (idx >= NB * 2) return;
    int n = idx >> 1, j = idx & 1;
    const float* row = h + ((size_t)n * 2 + j) * LL;
    float u[64];
    for (int o = 0; o < 64; ++o) {
        float s = bl1[o];
        for (int l = 0; l < LL; ++l) s += wl1[o * LL + l] * row[l];
        u[o] = act<2>(s);
    }
    float v3[3];
    for (int o = 0; o < 3; ++o) {
        float s = bl2[o];
        for (int k = 0; k < 64; ++k) s += wl2[o * 64 + k] * u[k];
        v3[o] = act<2>(s);
    }
    float s = bl3[0];
    for (int k = 0; k < 3; ++k) s += wl3[k] * v3[k];
    out[n * 2 + j] = s;   // output (N,1,2) flattened
}

// =====================================================================

extern "C" void kernel_launch(void* const* d_in, const int* in_sizes, int n_in,
                              void* d_out, int out_size, void* d_ws, size_t ws_size,
                              hipStream_t stream) {
    (void)in_sizes; (void)n_in; (void)out_size; (void)ws_size;

    const float* x    = (const float*)d_in[0];
    const float* w1d1 = (const float*)d_in[1];
    const float* b1d1 = (const float*)d_in[2];
    const float* bn1g = (const float*)d_in[3];
    const float* bn1b = (const float*)d_in[4];
    const float* w1d2 = (const float*)d_in[5];
    const float* b1d2 = (const float*)d_in[6];
    const float* bn2g = (const float*)d_in[7];
    const float* bn2b = (const float*)d_in[8];
    const float* gn1g = (const float*)d_in[9];
    const float* gn1b = (const float*)d_in[10];
    const float* ow1  = (const float*)d_in[11];
    const float* ob1  = (const float*)d_in[12];
    const float* gn2g = (const float*)d_in[13];
    const float* gn2b = (const float*)d_in[14];
    const float* ow2  = (const float*)d_in[15];
    const float* ob2  = (const float*)d_in[16];
    const float* gn3g = (const float*)d_in[17];
    const float* gn3b = (const float*)d_in[18];
    const float* nrmg = (const float*)d_in[19];
    const float* nrmb = (const float*)d_in[20];
    const float* wc1  = (const float*)d_in[21];
    const float* bc1  = (const float*)d_in[22];
    const float* wc2  = (const float*)d_in[23];
    const float* bc2  = (const float*)d_in[24];
    const float* wc3  = (const float*)d_in[25];
    const float* bc3  = (const float*)d_in[26];
    const float* wl1  = (const float*)d_in[27];
    const float* bl1  = (const float*)d_in[28];
    const float* wl2  = (const float*)d_in[29];
    const float* bl2  = (const float*)d_in[30];
    const float* wl3  = (const float*)d_in[31];
    const float* bl3  = (const float*)d_in[32];

    // ---- workspace carve-up (256B-aligned chunks) ----
    float* W = (float*)d_ws;
    size_t off = 0;
    auto take = [&](size_t nfl) { float* p = W + off; off += (nfl + 63) & ~(size_t)63; return p; };
    float*  y    = take(SZ);
    float*  kacc = take(SZ);
    float*  kcur = take(SZ);
    float*  ytmp = take(SZ);
    float*  tmp  = take(SZ);
    float*  h1   = take((size_t)NB * 64 * LL);   // frontend h1 / backend conv1 out
    float*  hb2  = take((size_t)NB * 16 * LL);
    float*  hb3  = take((size_t)NB * 2 * LL);
    __bf16* Xt   = (__bf16*)take(SZ / 2 + 64);   // bf16 [n][100][256]
    __bf16* Wk1  = (__bf16*)take(3 * 256 * 256 / 2);
    __bf16* Wk2  = (__bf16*)take(3 * 256 * 256 / 2);
    __bf16* Wkc  = (__bf16*)take(3 * 64 * 256 / 2);
    float*  tv1  = take(768);
    float*  tv2  = take(768);
    float*  sc1  = take(64);
    float*  sh1  = take(64);
    float*  sc2  = take(256);
    float*  sh2  = take(256);

    // ---- weight prep ----
    prep_ode_w<<<(3 * 256 * 256 + 255) / 256, 256, 0, stream>>>(ow1, Wk1, tv1);
    prep_ode_w<<<(3 * 256 * 256 + 255) / 256, 256, 0, stream>>>(ow2, Wk2, tv2);
    prep_wc<<<(3 * 64 * 256 + 255) / 256, 256, 0, stream>>>(wc1, Wkc, 64);

    // ---- frontend: conv1d(5->64) + BN relu + conv1d(64->256) + BN tanh -> y0 ----
    conv_plain<0><<<(NB * 64 * LL + 255) / 256, 256, 0, stream>>>(x, w1d1, b1d1, h1, 5, 64, 3, 0);
    bn_stats<<<64, 256, 0, stream>>>(h1, bn1g, bn1b, sc1, sh1, 64);
    bn_apply<1><<<(NB * 64 * LL + 255) / 256, 256, 0, stream>>>(h1, sc1, sh1, h1, 64);
    conv_plain<0><<<(SZ + 255) / 256, 256, 0, stream>>>(h1, w1d2, b1d2, tmp, 64, 256, 3, 0);
    bn_stats<<<256, 256, 0, stream>>>(tmp, bn2g, bn2b, sc2, sh2, 256);
    bn_apply<3><<<(SZ + 255) / 256, 256, 0, stream>>>(tmp, sc2, sh2, y, 256);

    // ---- ODE: RK4, host-unrolled ----
    const dim3 gconv(7, NB);
    auto odefunc = [&](const float* yin, float t, float* kout) {
        group_norm<1, 1><<<1024, 256, 0, stream>>>(yin, gn1g, gn1b, nullptr, Xt);
        wmma_conv<0, 16><<<gconv, 512, 0, stream>>>(Xt, Wk1, ob1, tv1, t, tmp);
        group_norm<2, 1><<<1024, 256, 0, stream>>>(tmp, gn2g, gn2b, nullptr, Xt);
        wmma_conv<0, 16><<<gconv, 512, 0, stream>>>(Xt, Wk2, ob2, tv2, t, tmp);
        group_norm<0, 0><<<1024, 256, 0, stream>>>(tmp, gn3g, gn3b, kout, nullptr);
    };

    const float dt = 0.25f;
    const int CB = SZ / 256;   // 25600 blocks, exact
    for (int s = 0; s < 4; ++s) {
        float t0 = s * dt;
        odefunc(y, t0, kcur);                                                        // k1
        comb3<<<CB, 256, 0, stream>>>(kacc, kcur, 1.f, kcur, 0.f, kcur, 0.f, SZ);    // kacc = k1
        comb3<<<CB, 256, 0, stream>>>(ytmp, y, 1.f, kcur, 0.5f * dt, kcur, 0.f, SZ); // y + dt/2 k1
        odefunc(ytmp, t0 + 0.5f * dt, kcur);                                         // k2
        comb3<<<CB, 256, 0, stream>>>(kacc, kacc, 1.f, kcur, 2.f, kcur, 0.f, SZ);
        comb3<<<CB, 256, 0, stream>>>(ytmp, y, 1.f, kcur, 0.5f * dt, kcur, 0.f, SZ);
        odefunc(ytmp, t0 + 0.5f * dt, kcur);                                         // k3
        comb3<<<CB, 256, 0, stream>>>(kacc, kacc, 1.f, kcur, 2.f, kcur, 0.f, SZ);
        comb3<<<CB, 256, 0, stream>>>(ytmp, y, 1.f, kcur, dt, kcur, 0.f, SZ);
        odefunc(ytmp, t0 + dt, kcur);                                                // k4
        comb3<<<CB, 256, 0, stream>>>(y, y, 1.f, kacc, dt / 6.f, kcur, dt / 6.f, SZ);
    }

    // ---- backend: GN relu -> wc1 (WMMA, elu) -> wc2 -> wc3 -> 3 linears ----
    group_norm<1, 1><<<1024, 256, 0, stream>>>(y, nrmg, nrmb, nullptr, Xt);
    wmma_conv<2, 4><<<dim3(7, NB), 128, 0, stream>>>(Xt, Wkc, bc1, nullptr, 0.f, h1);
    conv_plain<2><<<(NB * 16 * LL + 255) / 256, 256, 0, stream>>>(h1, wc2, bc2, hb2, 64, 16, 9, 3);
    conv_plain<2><<<(NB * 2 * LL + 255) / 256, 256, 0, stream>>>(hb2, wc3, bc3, hb3, 16, 2, 9, 3);
    head<<<2, 256, 0, stream>>>(hb3, wl1, bl1, wl2, bl2, wl3, bl3, (float*)d_out);
}
